// GAILPolicy_75539884802215
// MI455X (gfx1250) — matrix operand order repaired and checked by reading, not compiled, not run on hardware
//
#include <hip/hip_runtime.h>
#include <math.h>

typedef __attribute__((ext_vector_type(16))) _Float16 v16h;
typedef __attribute__((ext_vector_type(8)))  float    v8f;

#define T_N 65536
#define V_N 25
#define G_N 3
#define C_N 3
#define H_N 32

__device__ __forceinline__ float sigmoidf_(float v) { return 1.f / (1.f + __expf(-v)); }

// ---------------------------------------------------------------------------
// Kernel A: per 16-t tile, compute gout (T x 24) via WMMA + VALU, and the
// circle fit (cx, cy, cr) per timestep.  1 wave per block (barriers are NOPs).
// All WMMA operand gathers go through zero-padded LDS staging so fragment
// construction is branchless (no exec-mask diamonds, no per-element waits),
// and global traffic is coalesced float4 clauses.
// ---------------------------------------------------------------------------
__global__ __launch_bounds__(32)
void gail_features_kernel(const float* __restrict__ x,
                          const float* __restrict__ A,
                          const float* __restrict__ W1,
                          const float* __restrict__ b1,
                          const float* __restrict__ W2,
                          const float* __restrict__ b2,
                          float* __restrict__ gout,
                          float* __restrict__ cxA,
                          float* __restrict__ cyA,
                          float* __restrict__ crA)
{
    // v dimension padded 25 -> 32 and zero-filled: fragment gathers need no
    // bounds checks.
    __shared__ __align__(16) float xL[C_N][16][32][G_N]; // [c][t][v(pad32)][g]
    __shared__ __align__(16) float AL[32][32];           // A padded, zero-filled
    __shared__ float aggL[16][32][C_N];                  // [t][u][c] (c contiguous)
    __shared__ float pooledL[16][H_N];                   // [t][h]

    const int lane = threadIdx.x;
    const int t0   = blockIdx.x * 16;

    // ---- zero the padded staging buffers (vectorized) ----
    {
        float4 z4 = make_float4(0.f, 0.f, 0.f, 0.f);
        float4* xz = (float4*)&xL[0][0][0][0];           // 4608 floats / 4
        for (int i = lane; i < (C_N * 16 * 32 * G_N) / 4; i += 32) xz[i] = z4;
        float4* az = (float4*)&AL[0][0];                 // 1024 floats / 4
        for (int i = lane; i < (32 * 32) / 4; i += 32) az[i] = z4;
    }
    __syncthreads();

    // ---- stage x: per c the tile is 1200 CONTIGUOUS floats (16t x 25v x 3g) ----
    for (int c = 0; c < C_N; ++c) {
        const float4* src = (const float4*)(x + ((size_t)c * T_N + t0) * (V_N * G_N));
        float* dst = &xL[c][0][0][0];
        for (int i = lane; i < (16 * V_N * G_N) / 4; i += 32) {   // 300 float4
            float4 w = src[i];
            int e = i * 4;
            float vv[4] = { w.x, w.y, w.z, w.w };
#pragma unroll
            for (int q = 0; q < 4; ++q) {
                int ee = e + q;
                int tt = ee / (V_N * G_N);           // /75
                int rr = ee - tt * (V_N * G_N);      // v*3+g < 75
                dst[tt * (32 * G_N) + rr] = vv[q];   // padded row stride 96
            }
        }
    }
    // ---- stage A (25x25) into padded AL ----
    for (int i = lane; i < V_N * V_N; i += 32) {
        int r  = i / V_N;
        int cc = i - r * V_N;
        AL[r][cc] = A[i];
    }
    __syncthreads();

    // ---- B fragments from AL: f16 B 32x16 layout ----
    // lanes 0-15 hold K pairs {0..7,16..23}; lanes 16-31 K+8 (matches A).
    const int n   = lane & 15;
    const int kbB = (lane < 16) ? 0 : 8;
    v16h bf0, bf1;
#pragma unroll
    for (int j = 0; j < 8; ++j) {
#pragma unroll
        for (int p = 0; p < 2; ++p) {
            int k = (j < 4) ? (kbB + 2 * j + p) : (16 + kbB + 2 * (j - 4) + p);  // padded: k<32 always valid
            bf0[2 * j + p] = (_Float16)AL[k][n];
            bf1[2 * j + p] = (_Float16)AL[k][n + 16];
        }
    }

    // f16 A 16x32 layout: lanes 0-15 rows M, K pairs {0..7,16..23}; lanes 16-31 K+8.
    const int m   = lane & 15;
    const int kbA = (lane < 16) ? 0 : 8;

    for (int g = 0; g < G_N; ++g) {
#pragma unroll
        for (int c = 0; c < C_N; ++c) {
            v16h af;
#pragma unroll
            for (int j = 0; j < 8; ++j) {
#pragma unroll
                for (int p = 0; p < 2; ++p) {
                    int k = (j < 4) ? (kbA + 2 * j + p)
                                    : (16 + kbA + 2 * (j - 4) + p);  // v index (<32)
                    af[2 * j + p] = (_Float16)xL[c][m][k][g];
                }
            }
            v8f acc0 = {}, acc1 = {};
            acc0 = __builtin_amdgcn_wmma_f32_16x16x32_f16(false, af, false, bf0,
                                                          (short)0, acc0, false, false);
            acc1 = __builtin_amdgcn_wmma_f32_16x16x32_f16(false, af, false, bf1,
                                                          (short)0, acc1, false, false);
            // D layout: lanes 0-15 -> M=r, N=lane; lanes 16-31 -> M=r+8, N=lane-16
#pragma unroll
            for (int r = 0; r < 8; ++r) {
                int M = (lane < 16) ? r : (r + 8);
                aggL[M][n][c]      = acc0[r];
                aggL[M][n + 16][c] = acc1[r];
            }
        }
        __syncthreads();

        // hid = relu(W1 @ agg + b1); pooled = mean_u.  lane = h.
        {
            const int h  = lane;
            const float w0 = W1[h * C_N + 0];
            const float w1 = W1[h * C_N + 1];
            const float w2 = W1[h * C_N + 2];
            const float bb = b1[h];
            for (int tt = 0; tt < 16; ++tt) {
                float s = 0.f;
#pragma unroll
                for (int u = 0; u < V_N; ++u) {
                    const float* ap = &aggL[tt][u][0];   // 3 contiguous floats
                    float vv = fmaf(w0, ap[0], fmaf(w1, ap[1], fmaf(w2, ap[2], bb)));
                    s += fmaxf(vv, 0.f);
                }
                pooledL[tt][h] = s * (1.f / 25.f);
            }
        }
        __syncthreads();

        // gout = W2 @ pooled + b2.  lane = t (0..15).
        if (lane < 16) {
            const int tt = lane;
#pragma unroll
            for (int o = 0; o < 8; ++o) {
                float s = b2[o];
#pragma unroll
                for (int h2 = 0; h2 < H_N; ++h2)
                    s = fmaf(W2[o * H_N + h2], pooledL[tt][h2], s);
                gout[(size_t)(t0 + tt) * 24 + g * 8 + o] = s;
            }
        }
        __syncthreads();
    }

    // --- circle fit from joint 5, channels 0 (x) and 2 (y); read from LDS ---
    if (lane < 16) {
        const int tt = t0 + lane;
        float px0 = xL[0][lane][5][0];
        float px1 = xL[0][lane][5][1];
        float px2 = xL[0][lane][5][2];
        float py0 = xL[2][lane][5][0];
        float py1 = xL[2][lane][5][1];
        float py2 = xL[2][lane][5][2];
        float x21 = px1 - px0, y21 = py1 - py0;
        float x32 = px2 - px1, y32 = py2 - py1;
        float xy21 = px1 * px1 - px0 * px0 + py1 * py1 - py0 * py0;
        float xy32 = px2 * px2 - px1 * px1 + py2 * py2 - py1 * py1;
        float cy = (x32 * xy21 - x21 * xy32) * 0.5f * (y21 * x32 - y32 * x21);
        float cx = (xy21 - 2.f * cy * y21) / (2.f * x21);
        float cr = sqrtf((px0 - cx) * (px0 - cx) + (py0 - cy) * (py0 - cy));
        cxA[tt] = cx; cyA[tt] = cy; crA[tt] = cr;
    }
}

// ---------------------------------------------------------------------------
// Kernel B: the strictly serial 65535-step scan.  One wave32; lane = gate/row
// index; cross-lane traffic via shuffles.  All weights live in VGPRs.
// ---------------------------------------------------------------------------
__global__ __launch_bounds__(32)
void gail_scan_kernel(const float* __restrict__ x0,
                      const float* __restrict__ Wih0, const float* __restrict__ Whh0,
                      const float* __restrict__ bih0, const float* __restrict__ bhh0,
                      const float* __restrict__ Wih1, const float* __restrict__ Whh1,
                      const float* __restrict__ bih1, const float* __restrict__ bhh1,
                      const float* __restrict__ Wfc1, const float* __restrict__ bfc1,
                      const float* __restrict__ Wfc2, const float* __restrict__ bfc2,
                      const float* __restrict__ Wd1,  const float* __restrict__ bd1,
                      const float* __restrict__ Wd2,  const float* __restrict__ bd2,
                      const float* __restrict__ gout,
                      const float* __restrict__ cxA, const float* __restrict__ cyA,
                      const float* __restrict__ crA,
                      float* __restrict__ states_out, float* __restrict__ traj_out,
                      float* __restrict__ rewards_out, int* __restrict__ masks_out)
{
    const int l  = threadIdx.x;
    const int jj = l & 7;

    // ---- weights pinned in VGPRs ----
    float wi0a = Wih0[l * 2 + 0], wi0b = Wih0[l * 2 + 1];
    float wh0[8], wi1[8], wh1[8];
#pragma unroll
    for (int j = 0; j < 8; ++j) {
        wh0[j] = Whh0[l * 8 + j];
        wi1[j] = Wih1[l * 8 + j];
        wh1[j] = Whh1[l * 8 + j];
    }
    float bg0 = bih0[l] + bhh0[l];
    float bg1 = bih1[l] + bhh1[l];
    float wd1a[32], wd1b[32];
#pragma unroll
    for (int k = 0; k < 32; ++k) { wd1a[k] = Wd1[l * 32 + k]; wd1b[k] = Wd1[(l + 32) * 32 + k]; }
    float bd1a = bd1[l], bd1b = bd1[l + 32];
    float wd2a = Wd2[l], wd2b = Wd2[l + 32];
    float wf1[32]; float bf1v = 0.f;
    if (l < 8) {
        bf1v = bfc1[l];
#pragma unroll
        for (int k = 0; k < 32; ++k) wf1[k] = Wfc1[l * 32 + k];
    } else {
#pragma unroll
        for (int k = 0; k < 32; ++k) wf1[k] = 0.f;
    }
    float wf2[16];
#pragma unroll
    for (int j = 0; j < 16; ++j) wf2[j] = Wfc2[j];
    float bf2x = bfc2[0], bf2y = bfc2[1];
    float bd2v = bd2[0];

    // ---- state (replicated / lane-resident) ----
    float h0all[8], h1all[8];
#pragma unroll
    for (int j = 0; j < 8; ++j) { h0all[j] = 0.f; h1all[j] = 0.f; }
    float cm0 = 0.f, cm1 = 0.f;            // this lane's c element (index jj)
    float pt0 = x0[0], pt1 = x0[1];

    if (l == 0) { traj_out[0] = pt0; traj_out[1] = pt1; }

    for (int t = 0; t < T_N - 1; ++t) {
        // --- LSTM cell 0: lane computes gate row l ---
        float gv = fmaf(wi0a, pt0, fmaf(wi0b, pt1, bg0));
#pragma unroll
        for (int j = 0; j < 8; ++j) gv = fmaf(wh0[j], h0all[j], gv);
        float iv = __shfl(gv, jj);
        float fv = __shfl(gv, jj + 8);
        float gg = __shfl(gv, jj + 16);
        float ov = __shfl(gv, jj + 24);
        float c2  = sigmoidf_(fv) * cm0 + sigmoidf_(iv) * tanhf(gg);
        float hm0 = sigmoidf_(ov) * tanhf(c2);
        cm0 = c2;
#pragma unroll
        for (int j = 0; j < 8; ++j) h0all[j] = __shfl(hm0, j);

        // --- LSTM cell 1 ---
        float gv1 = bg1;
#pragma unroll
        for (int j = 0; j < 8; ++j) gv1 = fmaf(wi1[j], h0all[j], gv1);
#pragma unroll
        for (int j = 0; j < 8; ++j) gv1 = fmaf(wh1[j], h1all[j], gv1);
        float iv1 = __shfl(gv1, jj);
        float fv1 = __shfl(gv1, jj + 8);
        float gg1 = __shfl(gv1, jj + 16);
        float ov1 = __shfl(gv1, jj + 24);
        float c21 = sigmoidf_(fv1) * cm1 + sigmoidf_(iv1) * tanhf(gg1);
        float hm1 = sigmoidf_(ov1) * tanhf(c21);
        cm1 = c21;
#pragma unroll
        for (int j = 0; j < 8; ++j) h1all[j] = __shfl(hm1, j);

        // --- state[32] = [h1(8), gout[t](24)], lane l holds state[l] ---
        float sme = (l < 8) ? hm1 : gout[(size_t)t * 24 + (l - 8)];
        float st[32];
#pragma unroll
        for (int k = 0; k < 32; ++k) st[k] = __shfl(sme, k);

        // --- discriminator: reward = sigmoid(Wd2 @ tanh(Wd1 @ state + bd1) + bd2)
        float d1a = bd1a, d1b = bd1b;
#pragma unroll
        for (int k = 0; k < 32; ++k) {
            d1a = fmaf(wd1a[k], st[k], d1a);
            d1b = fmaf(wd1b[k], st[k], d1b);
        }
        float part = wd2a * tanhf(d1a) + wd2b * tanhf(d1b);
#pragma unroll
        for (int off = 16; off > 0; off >>= 1) part += __shfl_xor(part, off);
        float reward = sigmoidf_(part + bd2v);

        // --- policy head: nxt = Wfc2 @ (Wfc1 @ state + bfc1) + bfc2 ---
        float f1 = bf1v;
#pragma unroll
        for (int k = 0; k < 32; ++k) f1 = fmaf(wf1[k], st[k], f1);
        float f1a[8];
#pragma unroll
        for (int j = 0; j < 8; ++j) f1a[j] = __shfl(f1, j);
        float nx = bf2x, ny = bf2y;
#pragma unroll
        for (int j = 0; j < 8; ++j) {
            nx = fmaf(wf2[j],     f1a[j], nx);
            ny = fmaf(wf2[8 + j], f1a[j], ny);
        }

        float dx = nx - cxA[t], dy = ny - cyA[t];
        float dis = sqrtf(dx * dx + dy * dy);
        int msk = (dis < crA[t]) ? 1 : 0;

        // --- outputs ---
        states_out[(size_t)t * 32 + l] = sme;
        if (l == 0) {
            traj_out[(size_t)(t + 1) * 2 + 0] = nx;
            traj_out[(size_t)(t + 1) * 2 + 1] = ny;
            rewards_out[t] = reward;
            masks_out[t]   = msk;
        }
        pt0 = nx; pt1 = ny;
    }
}

extern "C" void kernel_launch(void* const* d_in, const int* in_sizes, int n_in,
                              void* d_out, int out_size, void* d_ws, size_t ws_size,
                              hipStream_t stream) {
    (void)in_sizes; (void)n_in; (void)out_size; (void)ws_size;
    const float* x    = (const float*)d_in[0];
    const float* x0   = (const float*)d_in[1];
    const float* A    = (const float*)d_in[2];
    const float* W1   = (const float*)d_in[3];
    const float* b1   = (const float*)d_in[4];
    const float* W2   = (const float*)d_in[5];
    const float* b2   = (const float*)d_in[6];
    const float* Wih0 = (const float*)d_in[7];
    const float* Whh0 = (const float*)d_in[8];
    const float* bih0 = (const float*)d_in[9];
    const float* bhh0 = (const float*)d_in[10];
    const float* Wih1 = (const float*)d_in[11];
    const float* Whh1 = (const float*)d_in[12];
    const float* bih1 = (const float*)d_in[13];
    const float* bhh1 = (const float*)d_in[14];
    const float* Wfc1 = (const float*)d_in[15];
    const float* bfc1 = (const float*)d_in[16];
    const float* Wfc2 = (const float*)d_in[17];
    const float* bfc2 = (const float*)d_in[18];
    const float* Wd1  = (const float*)d_in[19];
    const float* bd1  = (const float*)d_in[20];
    const float* Wd2  = (const float*)d_in[21];
    const float* bd2  = (const float*)d_in[22];

    // workspace: gout (T*24) | cx (T) | cy (T) | cr (T)
    float* ws   = (float*)d_ws;
    float* gout = ws;
    float* cxA  = ws + (size_t)T_N * 24;
    float* cyA  = cxA + T_N;
    float* crA  = cyA + T_N;

    // output: states (T-1,32) | traj (T,2) | rewards (T-1) | masks (T-1, int32)
    float* out         = (float*)d_out;
    float* states_out  = out;
    float* traj_out    = states_out + (size_t)(T_N - 1) * 32;
    float* rewards_out = traj_out + (size_t)T_N * 2;
    int*   masks_out   = (int*)(rewards_out + (T_N - 1));

    gail_features_kernel<<<dim3(T_N / 16), dim3(32), 0, stream>>>(
        x, A, W1, b1, W2, b2, gout, cxA, cyA, crA);

    gail_scan_kernel<<<dim3(1), dim3(32), 0, stream>>>(
        x0, Wih0, Whh0, bih0, bhh0, Wih1, Whh1, bih1, bhh1,
        Wfc1, bfc1, Wfc2, bfc2, Wd1, bd1, Wd2, bd2,
        gout, cxA, cyA, crA, states_out, traj_out, rewards_out, masks_out);
}
